// AdditiveAttention_54752243090125
// MI455X (gfx1250) — matrix operand order repaired and checked by reading, not compiled, not run on hardware
//
#include <hip/hip_runtime.h>
#include <hip/hip_bf16.h>

// Shapes (from reference)
#define D_MODEL 512
#define ENC_C   1024   // 2*D_MODEL
#define BATCH   16
#define DEC_LEN 256
#define ENC_LEN 2048

typedef __attribute__((ext_vector_type(16))) __bf16       v16bf;
typedef __attribute__((ext_vector_type(8)))  float        v8f;
typedef __attribute__((ext_vector_type(8)))  unsigned int v8u;

union Frag16 { v8u u; v16bf h; };

__device__ __forceinline__ float wave_reduce_sum(float v) {
  #pragma unroll
  for (int off = 16; off > 0; off >>= 1)
    v += __shfl_down(v, off, 32);
  return v;
}

// ---------------------------------------------------------------------------
// K1: v_enc[c] = sum_m W_enc[c,m] * w_out[m]     (1024 rows of length 512)
// One wave32 per row; float4 loads; shfl reduce.
// ---------------------------------------------------------------------------
__global__ void k_proj(const float* __restrict__ W_enc,
                       const float* __restrict__ w_out,
                       float* __restrict__ v_enc) {
  int wid  = threadIdx.x >> 5;
  int lane = threadIdx.x & 31;
  int row  = blockIdx.x * 8 + wid;             // 0..1023
  if (row >= ENC_C) return;
  const float4* w = (const float4*)(W_enc + (long)row * D_MODEL);
  const float4* x = (const float4*)w_out;
  float s = 0.f;
  for (int j = lane; j < D_MODEL / 4; j += 32) {
    float4 a = w[j], b = x[j];
    s += a.x * b.x + a.y * b.y + a.z * b.z + a.w * b.w;
  }
  s = wave_reduce_sum(s);
  if (lane == 0) v_enc[row] = s;
}

// ---------------------------------------------------------------------------
// K2: score_e[b,e] = enc[b,e,:] . v_enc          (32768 rows of length 1024)
// One wave32 per row. This is the dominant HBM pass (134 MB of enc).
// ---------------------------------------------------------------------------
__global__ void k_score(const float* __restrict__ enc,
                        const float* __restrict__ v_enc,
                        float* __restrict__ score_e) {
  int wid  = threadIdx.x >> 5;
  int lane = threadIdx.x & 31;
  long row = (long)blockIdx.x * 8 + wid;       // b*ENC_LEN + e
  const float4* er = (const float4*)(enc + row * ENC_C);
  const float4* v  = (const float4*)v_enc;
  float s = 0.f;
  for (int j = lane; j < ENC_C / 4; j += 32) {
    float4 a = er[j], b = v[j];
    s += a.x * b.x + a.y * b.y + a.z * b.z + a.w * b.w;
  }
  s = wave_reduce_sum(s);
  if (lane == 0) score_e[row] = s;
}

// ---------------------------------------------------------------------------
// K3: attn_row[b,:] = softmax(score_e[b,:] + log(mask[b,:])) over ENC_LEN.
// (score_d and b_out cancel under softmax shift invariance.)
// One 256-thread block per batch.
// ---------------------------------------------------------------------------
__global__ void k_softmax(const float* __restrict__ score_e,
                          const float* __restrict__ enc_mask,
                          float* __restrict__ attn_row) {
  __shared__ float red[8];
  int b = blockIdx.x;
  int t = threadIdx.x;
  int wid = t >> 5, lane = t & 31;

  float v[8];
  float m = -3.402823466e38f;
  #pragma unroll
  for (int j = 0; j < 8; ++j) {
    int e = t + 256 * j;
    float s = score_e[(long)b * ENC_LEN + e] + __logf(enc_mask[(long)b * ENC_LEN + e]);
    v[j] = s;
    m = fmaxf(m, s);
  }
  #pragma unroll
  for (int off = 16; off > 0; off >>= 1) m = fmaxf(m, __shfl_down(m, off, 32));
  if (lane == 0) red[wid] = m;
  __syncthreads();
  m = red[0];
  #pragma unroll
  for (int i = 1; i < 8; ++i) m = fmaxf(m, red[i]);
  __syncthreads();

  float s = 0.f;
  #pragma unroll
  for (int j = 0; j < 8; ++j) { v[j] = __expf(v[j] - m); s += v[j]; }
  s = wave_reduce_sum(s);
  if (lane == 0) red[wid] = s;
  __syncthreads();
  s = 0.f;
  #pragma unroll
  for (int i = 0; i < 8; ++i) s += red[i];
  float inv = 1.0f / s;

  #pragma unroll
  for (int j = 0; j < 8; ++j)
    attn_row[(long)b * ENC_LEN + t + 256 * j] = v[j] * inv;
}

// ---------------------------------------------------------------------------
// K3b: attn_dist[b,d,:] = attn_row[b,:] for every d (broadcast write, 33.5 MB).
// One block per (b,d) row; coalesced float4 stores; source row stays in L2.
// ---------------------------------------------------------------------------
__global__ void k_bcast_attn(const float* __restrict__ attn_row,
                             float* __restrict__ out_attn) {
  int row = blockIdx.x;                        // b*DEC_LEN + d
  int b   = row >> 8;
  int t   = threadIdx.x;
  const float4* src = (const float4*)(attn_row + (long)b * ENC_LEN);
  float4* dst = (float4*)(out_attn + (long)row * ENC_LEN);
  dst[t]       = src[t];
  dst[t + 256] = src[t + 256];
}

// ---------------------------------------------------------------------------
// K4: context[b,d,c] = sum_e attn_row[b,e] * enc[b,e,c]  via WMMA.
// A-fragment (16x32 bf16): all 16 M-rows are the same attn slice, so one
// v_wmma_f32_16x16x32_bf16 tile yields the context slice for 16 channels;
// D-row 0 (lanes 0-15, acc[0]) is then broadcast over all DEC_LEN rows with
// coalesced float4 stores (tile parked in LDS first).
//
// Layouts (D per CDNA5 ISA):
//   A 16x32 bf16: lanes 0-15 halfs = K0..15, lanes 16-31 = K16..31 (same as B)
//   B 32x16 bf16: lane = column N%16; lanes 0-15 halfs = K0..15, lanes 16-31 = K16..31 (contiguous k, not the ISA table order)
//   D 16x16 f32 : VGPR0 lanes 0-15 -> M=0, lanes 16-31 -> M=8
//
// All global offsets kept in 32-bit (per-batch enc block = 8.4 MB) so the
// backend can use SGPR-base + VGPR-offset addressing with the 16 row loads
// folded into 24-bit instruction offsets (h*4096 B).
// ---------------------------------------------------------------------------
__global__ void k_context_wmma(const float* __restrict__ enc,
                               const float* __restrict__ attn_row,
                               float* __restrict__ out_ctx) {
  __shared__ unsigned short sA[ENC_LEN];       // attn row in bf16 (4 KB)
  __shared__ __align__(16) float sC[8][16];    // per-wave context tile
  int b = blockIdx.y;
  int t = threadIdx.x;

  for (int j = t; j < ENC_LEN; j += 256) {
    __bf16 h = (__bf16)attn_row[(long)b * ENC_LEN + j];
    sA[j] = __builtin_bit_cast(unsigned short, h);
  }
  __syncthreads();

  int wid  = t >> 5;
  int lane = t & 31;
  int c0   = (blockIdx.x * 8 + wid) * 16;      // 16-channel tile
  int cl   = c0 + (lane & 15);
  int hi   = (lane >= 16) ? 1 : 0;

  const unsigned int* sA32 = (const unsigned int*)sA;   // 2 bf16 per dword
  const float* encB = enc + (long)b * ENC_LEN * ENC_C;  // uniform SGPR base

  // Per-lane 32-bit element offset into encB for B-fragment column gather.
  unsigned boff = (unsigned)(hi * 16 * ENC_C + cl);

  v8f acc = {};
  for (int k = 0; k < ENC_LEN; k += 32) {
    // A fragment from LDS (identical across lanes of each half-wave)
    Frag16 a;
    int base = (k >> 1) + (hi ? 8 : 0);
    #pragma unroll
    for (int i = 0; i < 8; ++i) {
      a.u[i]     = sA32[base + i];             // K base+{0..15}
    }
    // B fragment: column cl, 16 rows at constant strides h*ENC_C (imm offsets)
    Frag16 bm;
    #pragma unroll
    for (int h = 0; h < 16; ++h)
      bm.h[h] = (__bf16)encB[boff + (unsigned)(h * ENC_C)];
    boff += 32u * ENC_C;

    acc = __builtin_amdgcn_wmma_f32_16x16x32_bf16(
        /*neg_a=*/false, a.h, /*neg_b=*/false, bm.h,
        /*c_mod=*/(short)0, acc, /*reuse_a=*/false, /*reuse_b=*/false);
  }

  // Park this wave's 16-float context slice in LDS (rows of D are identical;
  // take M=0 from lanes 0-15).
  if (lane < 16) sC[wid][lane] = acc[0];
  __syncthreads();

  // Broadcast over all DEC_LEN rows with float4 stores:
  // lane -> quad q (0..3) within row, sub-row dr (0..7); 8 rows per iteration.
  int q  = lane & 3;
  int dr = lane >> 2;
  float4 cval = *((const float4*)&sC[wid][4 * q]);
  float4* ctx4 = (float4*)(out_ctx + (long)b * DEC_LEN * ENC_C);  // uniform base
  unsigned idx4 = (unsigned)(dr * (ENC_C / 4) + (c0 >> 2) + q);
  for (int p = 0; p < DEC_LEN / 8; ++p) {
    ctx4[idx4] = cval;
    idx4 += 8u * (ENC_C / 4);
  }
}

// ---------------------------------------------------------------------------
extern "C" void kernel_launch(void* const* d_in, const int* in_sizes, int n_in,
                              void* d_out, int out_size, void* d_ws, size_t ws_size,
                              hipStream_t stream) {
  // setup_inputs order:
  // 0 decoder_states, 1 decoder_mask, 2 encoder_states, 3 encoder_mask,
  // 4 W_enc, 5 W_dec, 6 w_out, 7 b_out
  const float* enc      = (const float*)d_in[2];
  const float* enc_mask = (const float*)d_in[3];
  const float* W_enc    = (const float*)d_in[4];
  const float* w_out_v  = (const float*)d_in[6];
  // decoder_states / W_dec / b_out cancel under softmax shift-invariance.

  float* ws       = (float*)d_ws;
  float* v_enc    = ws;                          // 1024 f32
  float* score_e  = ws + 1024;                   // 16*2048 f32
  float* attn_row = ws + 1024 + BATCH * ENC_LEN; // 16*2048 f32

  float* out_ctx  = (float*)d_out;                                   // [16,256,1024]
  float* out_attn = out_ctx + (size_t)BATCH * DEC_LEN * ENC_C;       // [16,256,2048]

  k_proj        <<<ENC_C / 8, 256, 0, stream>>>(W_enc, w_out_v, v_enc);
  k_score       <<<(BATCH * ENC_LEN) / 8, 256, 0, stream>>>(enc, v_enc, score_e);
  k_softmax     <<<BATCH, 256, 0, stream>>>(score_e, enc_mask, attn_row);
  k_bcast_attn  <<<BATCH * DEC_LEN, 256, 0, stream>>>(attn_row, out_attn);
  k_context_wmma<<<dim3(ENC_C / 128, BATCH), 256, 0, stream>>>(enc, attn_row, out_ctx);
}
